// MultiHeadAttentionLayer_45174466020059
// MI455X (gfx1250) — matrix operand order repaired and checked by reading, not compiled, not run on hardware
//
#include <hip/hip_runtime.h>
#include <hip/hip_bf16.h>
#include <cstdint>
#include <cstddef>

typedef __attribute__((ext_vector_type(16))) __bf16 v16bf;
typedef __attribute__((ext_vector_type(8)))  float  v8f;
typedef __attribute__((ext_vector_type(4)))  int    i32x4;

#define BM 128
#define BN 128
#define KSTEP 64
#define LSTR 72   // padded LDS row stride (bf16): 144B rows, 16B-aligned, conflict-free frag reads

// ---- gfx1250 async global->LDS path (guarded; falls back to reg-prefetch) ----
#if defined(__has_builtin)
#if __has_builtin(__builtin_amdgcn_global_load_async_to_lds_b128) && \
    __has_builtin(__builtin_amdgcn_s_wait_asynccnt)
#define HAVE_ASYNC_LDS 1
#endif
#endif
#ifndef HAVE_ASYNC_LDS
#define HAVE_ASYNC_LDS 0
#endif

#if HAVE_ASYNC_LDS
typedef __attribute__((address_space(1))) i32x4 gvec_t;
typedef __attribute__((address_space(3))) i32x4 lvec_t;
__device__ __forceinline__ void async_copy_b128(const void* gsrc, void* ldst) {
  __builtin_amdgcn_global_load_async_to_lds_b128(
      (gvec_t*)gsrc, (lvec_t*)ldst, 0, 0);
}
__device__ __forceinline__ void async_wait_all() { __builtin_amdgcn_s_wait_asynccnt(0); }
#else
__device__ __forceinline__ void async_copy_b128(const void*, void*) {}
__device__ __forceinline__ void async_wait_all() {}
#endif
constexpr bool kUseAsync = (HAVE_ASYNC_LDS != 0);

__device__ __forceinline__ unsigned short f32_to_bf16_rne(float f) {
  union { float f; unsigned int u; } x; x.f = f;
  unsigned int u = x.u;
  unsigned int r = u + 0x7FFFu + ((u >> 16) & 1u);
  return (unsigned short)(r >> 16);
}
__device__ __forceinline__ unsigned int pack2(float a, float b) {
  return (unsigned int)f32_to_bf16_rne(a) | ((unsigned int)f32_to_bf16_rne(b) << 16);
}
struct P128x2 { uint4 a, b; };
__device__ __forceinline__ v16bf as_v16bf(uint4 a, uint4 b) {
  P128x2 p; p.a = a; p.b = b;
  return __builtin_bit_cast(v16bf, p);
}

// Generic double-buffered bf16-WMMA GEMM: C[M,N] = A[M,K] @ B + bias
//  A_F32:    A is f32 in global (bf16-converted while staging), else bf16 (async-copied if available)
//  B_KMAJOR: B stored [N][K] (K contiguous, async-copied if available) else [K][N] (LDS transpose)
//  OUT_BF16: write bf16 (else f32)
// Batched over gridDim.z: z -> (b = z/nH, h = z%nH) with per-operand element strides.
template<bool A_F32, bool B_KMAJOR, bool OUT_BF16, bool HAS_BIAS>
__global__ __launch_bounds__(256)
void gemm_wmma(const void* __restrict__ Aptr, long long lda, long long sAb, long long sAh,
               const unsigned short* __restrict__ Bptr, long long ldb, long long sBb, long long sBh,
               void* __restrict__ Cptr, long long ldc, long long sCb, long long sCh,
               const float* __restrict__ bias,
               int M, int N, int K, int nH)
{
  __shared__ unsigned short ldsA[2][BM * LSTR];
  __shared__ unsigned short ldsB[2][BN * LSTR];

  const int z  = blockIdx.z;
  const int bb = z / nH, hh = z % nH;
  const long long aoff = (long long)bb * sAb + (long long)hh * sAh;
  const long long boff = (long long)bb * sBb + (long long)hh * sBh;
  const long long coff = (long long)bb * sCb + (long long)hh * sCh;

  const float*          Af = (const float*)Aptr;
  const unsigned short* Ab = (const unsigned short*)Aptr;

  const int bm = blockIdx.y * BM;
  const int bn = blockIdx.x * BN;
  const int tid  = threadIdx.x;
  const int lane = tid & 31;
  const int wave = tid >> 5;
  const int wm = wave & 3;    // 4 waves along M (32 rows each)
  const int wn = wave >> 2;   // 2 waves along N (64 cols each)
  const int lrow = lane & 15;
  const int lhi  = lane >> 4;

  v8f acc[2][4];
  #pragma unroll
  for (int i = 0; i < 2; ++i)
    #pragma unroll
    for (int j = 0; j < 4; ++j)
      #pragma unroll
      for (int r = 0; r < 8; ++r) acc[i][j][r] = 0.0f;

  // staging geometry: each thread owns 32 contiguous K-elems of one row (A / K-major B)
  const int r2     = tid >> 1;           // tile row 0..127
  const int half32 = (tid & 1) * 32;     // K-half within 64-wide slice
  const int kcol   = tid & 63;           // transpose path: k within slice
  const int nbase  = (tid >> 6) * 32;    // transpose path: 32-wide n chunk

  const float*          aSrcF = Af + aoff + (long long)(bm + r2) * lda + half32;
  const unsigned short* aSrcH = Ab + aoff + (long long)(bm + r2) * lda + half32;
  const unsigned short* bSrcK = Bptr + boff + (long long)(bn + r2) * ldb + half32;
  const unsigned short* bSrcT = Bptr + boff + (long long)kcol * ldb + bn + nbase;

  uint4 aReg[8];
  uint4 bReg[4];

  auto issueA = [&](int kt, int nbuf) {
    if constexpr (A_F32) {
      const float4* s = (const float4*)(aSrcF + kt);
      #pragma unroll
      for (int i = 0; i < 8; ++i) aReg[i] = __builtin_bit_cast(uint4, s[i]);
    } else {
      const uint4* s = (const uint4*)(aSrcH + kt);
      if constexpr (kUseAsync) {
        unsigned short* d = &ldsA[nbuf][r2 * LSTR + half32];
        #pragma unroll
        for (int i = 0; i < 4; ++i) async_copy_b128(s + i, d + i * 8);
      } else {
        #pragma unroll
        for (int i = 0; i < 4; ++i) aReg[i] = s[i];
      }
    }
  };
  auto writeA = [&](int nbuf) {
    uint4* dst = (uint4*)&ldsA[nbuf][r2 * LSTR + half32];
    if constexpr (A_F32) {
      #pragma unroll
      for (int i = 0; i < 4; ++i) {
        float4 f0 = __builtin_bit_cast(float4, aReg[2 * i]);
        float4 f1 = __builtin_bit_cast(float4, aReg[2 * i + 1]);
        dst[i] = make_uint4(pack2(f0.x, f0.y), pack2(f0.z, f0.w),
                            pack2(f1.x, f1.y), pack2(f1.z, f1.w));
      }
    } else if constexpr (!kUseAsync) {
      #pragma unroll
      for (int i = 0; i < 4; ++i) dst[i] = aReg[i];
    }
  };
  auto issueB = [&](int kt, int nbuf) {
    if constexpr (B_KMAJOR) {
      const uint4* s = (const uint4*)(bSrcK + kt);
      if constexpr (kUseAsync) {
        unsigned short* d = &ldsB[nbuf][r2 * LSTR + half32];
        #pragma unroll
        for (int i = 0; i < 4; ++i) async_copy_b128(s + i, d + i * 8);
      } else {
        #pragma unroll
        for (int i = 0; i < 4; ++i) bReg[i] = s[i];
      }
    } else {
      const uint4* s = (const uint4*)(bSrcT + (long long)kt * ldb);
      #pragma unroll
      for (int i = 0; i < 4; ++i) bReg[i] = s[i];
    }
  };
  auto writeB = [&](int nbuf) {
    if constexpr (B_KMAJOR) {
      if constexpr (!kUseAsync) {
        uint4* dst = (uint4*)&ldsB[nbuf][r2 * LSTR + half32];
        #pragma unroll
        for (int i = 0; i < 4; ++i) dst[i] = bReg[i];
      }
    } else {
      unsigned short e[32];
      *(uint4*)&e[0]  = bReg[0];
      *(uint4*)&e[8]  = bReg[1];
      *(uint4*)&e[16] = bReg[2];
      *(uint4*)&e[24] = bReg[3];
      #pragma unroll
      for (int i = 0; i < 32; ++i)
        ldsB[nbuf][(nbase + i) * LSTR + kcol] = e[i];
    }
  };
  auto compute = [&](int cbuf) {
    #pragma unroll
    for (int ks = 0; ks < KSTEP; ks += 32) {
      v16bf af[2], bfg[4];
      #pragma unroll
      for (int i = 0; i < 2; ++i) {
        // A frag (ISA 7.12.2): lane<16 -> K ks+0..7 & ks+16..23 ; lane>=16 -> K ks+8..15 & ks+24..31
        const unsigned short* pa = &ldsA[cbuf][(wm * 32 + i * 16 + lrow) * LSTR + ks + lhi * 8];
        af[i] = as_v16bf(*(const uint4*)pa, *(const uint4*)(pa + 16));
      }
      #pragma unroll
      for (int j = 0; j < 4; ++j) {
        // B frag: lane<16 -> N=lane, K ks+0..7 & ks+16..23 ; lane>=16 -> N=lane-16, K ks+8..15 & ks+24..31
        const unsigned short* pb = &ldsB[cbuf][(wn * 64 + j * 16 + lrow) * LSTR + ks + lhi * 8];
        bfg[j] = as_v16bf(*(const uint4*)pb, *(const uint4*)(pb + 16));
      }
      #pragma unroll
      for (int i = 0; i < 2; ++i)
        #pragma unroll
        for (int j = 0; j < 4; ++j)
          acc[i][j] = __builtin_amdgcn_wmma_f32_16x16x32_bf16(
              false, af[i], false, bfg[j], (short)0, acc[i][j], false, false);
    }
  };

  // ---- pipelined main loop (ping-pong buffers, one barrier per K-slice) ----
  issueA(0, 0); issueB(0, 0);
  writeA(0);    writeB(0);
  async_wait_all();
  __syncthreads();

  int buf = 0;
  for (int kt = 0; kt < K; kt += KSTEP) {
    const int  nxt  = kt + KSTEP;
    const bool more = nxt < K;
    if (more) { issueA(nxt, buf ^ 1); issueB(nxt, buf ^ 1); }
    compute(buf);
    if (more) {
      writeA(buf ^ 1); writeB(buf ^ 1);
      async_wait_all();
      __syncthreads();
    }
    buf ^= 1;
  }

  // ---- store (C/D layout: lane<16 N=lane M=r ; lane>=16 N=lane-16 M=r+8) ----
  float*          Cf = (float*)Cptr;
  unsigned short* Cb = (unsigned short*)Cptr;
  #pragma unroll
  for (int i = 0; i < 2; ++i) {
    #pragma unroll
    for (int j = 0; j < 4; ++j) {
      const int n = bn + wn * 64 + j * 16 + lrow;
      const float bz = HAS_BIAS ? bias[n] : 0.0f;
      #pragma unroll
      for (int r = 0; r < 8; ++r) {
        const int m = bm + wm * 32 + i * 16 + r + 8 * lhi;
        const float v = acc[i][j][r] + bz;
        if constexpr (OUT_BF16)
          Cb[coff + (long long)m * ldc + n] = f32_to_bf16_rne(v);
        else
          Cf[coff + (long long)m * ldc + n] = v;
      }
    }
  }
}

// Fold the group-sum into Wq: WqsT[j'][d] = sum_g Wq[d][(kvh*4+g)*128 + j], bf16, transposed.
__global__ void prep_wqs_kernel(const float* __restrict__ Wq, const float* __restrict__ bq,
                                unsigned short* __restrict__ WqsT, float* __restrict__ bqs) {
  int idx = blockIdx.x * 256 + threadIdx.x;
  if (idx < 512 * 2048) {
    int jp = idx >> 11;
    int d  = idx & 2047;
    int kvh = jp >> 7, j = jp & 127;
    float s = 0.f;
    #pragma unroll
    for (int g = 0; g < 4; ++g)
      s += Wq[(size_t)d * 2048 + kvh * 512 + g * 128 + j];
    WqsT[idx] = f32_to_bf16_rne(s);
  }
  if (idx < 512) {
    int kvh = idx >> 7, j = idx & 127;
    float s = 0.f;
    #pragma unroll
    for (int g = 0; g < 4; ++g) s += bq[kvh * 512 + g * 128 + j];
    bqs[idx] = s;
  }
}

// out[c][r] = bf16(in[r][c]) : produce K-major bf16 weights
__global__ void transpose_bf16_kernel(const float* __restrict__ in, unsigned short* __restrict__ out,
                                      int R, int C) {
  int idx = blockIdx.x * 256 + threadIdx.x;
  if (idx >= R * C) return;
  int c = idx / R, r = idx - c * R;
  out[idx] = f32_to_bf16_rne(in[(size_t)r * C + c]);
}

// In-place softmax over rows of 2048 with pre-scale (1/sqrt(128))
__global__ __launch_bounds__(256) void softmax2048(float* __restrict__ attn, float inv) {
  __shared__ float red[256];
  float* p = attn + (size_t)blockIdx.x * 2048;
  const int t = threadIdx.x;
  float v[8];
  float mx = -3.402823466e38f;
  #pragma unroll
  for (int i = 0; i < 8; ++i) { v[i] = p[t + 256 * i]; mx = fmaxf(mx, v[i]); }
  red[t] = mx; __syncthreads();
  for (int s = 128; s > 0; s >>= 1) { if (t < s) red[t] = fmaxf(red[t], red[t + s]); __syncthreads(); }
  mx = red[0]; __syncthreads();
  float sum = 0.f;
  #pragma unroll
  for (int i = 0; i < 8; ++i) { v[i] = __expf((v[i] - mx) * inv); sum += v[i]; }
  red[t] = sum; __syncthreads();
  for (int s = 128; s > 0; s >>= 1) { if (t < s) red[t] += red[t + s]; __syncthreads(); }
  const float rs = 1.0f / red[0];
  #pragma unroll
  for (int i = 0; i < 8; ++i) p[t + 256 * i] = v[i] * rs;
}

extern "C" void kernel_launch(void* const* d_in, const int* in_sizes, int n_in,
                              void* d_out, int out_size, void* d_ws, size_t ws_size,
                              hipStream_t stream) {
  (void)in_sizes; (void)n_in; (void)out_size; (void)ws_size;
  const float* query = (const float*)d_in[0];
  const float* key   = (const float*)d_in[1];
  const float* value = (const float*)d_in[2];
  const float* Wq = (const float*)d_in[3];
  const float* bq = (const float*)d_in[4];
  const float* Wk = (const float*)d_in[5];
  const float* bk = (const float*)d_in[6];
  const float* Wv = (const float*)d_in[7];
  const float* bv = (const float*)d_in[8];
  const float* Wo = (const float*)d_in[9];
  const float* bo = (const float*)d_in[10];

  const int Bb = 4, Nn = 2048, Dd = 2048, KVD = 512, nKVH = 4;

  char* ws = (char*)d_ws;
  size_t off = 0;
  auto alloc = [&](size_t bytes) { char* p = ws + off; off += (bytes + 255) & ~(size_t)255; return p; };
  unsigned short* WqsT = (unsigned short*)alloc((size_t)512 * 2048 * 2);
  unsigned short* WkT  = (unsigned short*)alloc((size_t)512 * 2048 * 2);
  unsigned short* WvT  = (unsigned short*)alloc((size_t)512 * 2048 * 2);
  unsigned short* WoT  = (unsigned short*)alloc((size_t)2048 * 512 * 2);
  float*          bqs  = (float*)alloc(512 * 4);
  unsigned short* Qs = (unsigned short*)alloc((size_t)8192 * 512 * 2);
  unsigned short* Kb = (unsigned short*)alloc((size_t)8192 * 512 * 2);
  unsigned short* Vb = (unsigned short*)alloc((size_t)8192 * 512 * 2);
  unsigned short* Xb = (unsigned short*)alloc((size_t)8192 * 512 * 2);

  float* outX = (float*)d_out;                              // [4,2048,2048]
  float* attn = outX + (size_t)Bb * Nn * Dd;                // [4,4,2048,2048]

  // ---- weight prep (bf16, K-major) ----
  prep_wqs_kernel<<<(512 * 2048 + 255) / 256, 256, 0, stream>>>(Wq, bq, WqsT, bqs);
  transpose_bf16_kernel<<<(2048 * 512 + 255) / 256, 256, 0, stream>>>(Wk, WkT, 2048, 512);
  transpose_bf16_kernel<<<(2048 * 512 + 255) / 256, 256, 0, stream>>>(Wv, WvT, 2048, 512);
  transpose_bf16_kernel<<<(512 * 2048 + 255) / 256, 256, 0, stream>>>(Wo, WoT, 512, 2048);

  // ---- projections: [8192,2048] @ [2048,512] + bias -> bf16 [8192,512] ----
  {
    dim3 grid(512 / BN, 8192 / BM, 1);
    gemm_wmma<true, true, true, true><<<grid, 256, 0, stream>>>(
        query, Dd, 0, 0, WqsT, Dd, 0, 0, Qs, KVD, 0, 0, bqs, 8192, 512, 2048, 1);
    gemm_wmma<true, true, true, true><<<grid, 256, 0, stream>>>(
        key, Dd, 0, 0, WkT, Dd, 0, 0, Kb, KVD, 0, 0, bk, 8192, 512, 2048, 1);
    gemm_wmma<true, true, true, true><<<grid, 256, 0, stream>>>(
        value, Dd, 0, 0, WvT, Dd, 0, 0, Vb, KVD, 0, 0, bv, 8192, 512, 2048, 1);
  }
  // ---- energy[b,h] = Qs_h @ K_h^T -> f32 into d_out attention region ----
  {
    dim3 grid(2048 / BN, 2048 / BM, 16);
    gemm_wmma<false, true, false, false><<<grid, 256, 0, stream>>>(
        Qs, KVD, (long long)Nn * KVD, 128,
        Kb, KVD, (long long)Nn * KVD, 128,
        attn, Nn, (long long)nKVH * Nn * Nn, (long long)Nn * Nn,
        nullptr, 2048, 2048, 128, nKVH);
  }
  // ---- softmax(energy / sqrt(128)) in place ----
  softmax2048<<<Bb * nKVH * Nn, 256, 0, stream>>>(attn, 0.0883883476483184f);
  // ---- x[b,h] = attn[b,h] @ V_h -> bf16 [b, n, h*128+d] ----
  {
    dim3 grid(1, 2048 / BM, 16);
    gemm_wmma<true, false, true, false><<<grid, 256, 0, stream>>>(
        attn, Nn, (long long)nKVH * Nn * Nn, (long long)Nn * Nn,
        Vb, KVD, (long long)Nn * KVD, 128,
        Xb, KVD, (long long)Nn * KVD, 128,
        nullptr, 2048, 128, 2048, nKVH);
  }
  // ---- out = x @ Wo + bo -> f32 [4,2048,2048] ----
  {
    dim3 grid(2048 / BN, 8192 / BM, 1);
    gemm_wmma<false, true, false, true><<<grid, 256, 0, stream>>>(
        Xb, KVD, 0, 0, WoT, KVD, 0, 0, outX, Dd, 0, 0, bo, 8192, 2048, 512, 1);
  }
}
